// PtrExtractSummGAT_51539607552923
// MI455X (gfx1250) — matrix operand order repaired and checked by reading, not compiled, not run on hardware
//
#include <hip/hip_runtime.h>
#include <hip/hip_bf16.h>
#include <cstddef>

typedef __bf16 bf16_t;
typedef __attribute__((ext_vector_type(16))) __bf16 v16bf;
typedef __attribute__((ext_vector_type(8)))  __bf16 v8bf;
typedef __attribute__((ext_vector_type(8)))  float  v8f;

#define BB   128
#define NP1  257
#define DD   512
#define HH   512
#define T9   9
#define G4H  2048

// Async memory->LDS path (CDNA5 GLOBAL_LOAD_ASYNC_TO_LDS_*, ASYNCcnt).
#if defined(__has_builtin)
#if __has_builtin(__builtin_amdgcn_global_load_async_to_lds_b128) && \
    __has_builtin(__builtin_amdgcn_s_wait_asynccnt)
#define HAS_ASYNC_LDS 1
#endif
#endif

#ifdef HAS_ASYNC_LDS
typedef int v4i_vs __attribute__((vector_size(4 * sizeof(int))));
#define AS1 __attribute__((address_space(1)))
#define AS3 __attribute__((address_space(3)))
__device__ __forceinline__ void async_copy16(const bf16_t* g, bf16_t* l) {
  __builtin_amdgcn_global_load_async_to_lds_b128(
      (AS1 v4i_vs*)(v4i_vs*)(void*)const_cast<bf16_t*>(g),
      (AS3 v4i_vs*)(v4i_vs*)(void*)l,
      /*offset=*/0, /*cpol=*/0);
}
#endif

// ---------------------------------------------------------------------------
// Generic bf16 WMMA GEMM: C[M,N] = A[M,K] * Bt[N,K]^T (+ bias[n] optional)
// Block = 256 threads (8 waves). Tile = 128 (M) x 64 (N), K stepped by 32.
// Wave w computes rows [w*16, w*16+16) x all 64 cols (4 accumulators).
// Requires: M % 128 == 0, N % 64 == 0, K % 32 == 0.
// ---------------------------------------------------------------------------
__global__ __launch_bounds__(256) void gemm_bf16_kernel(
    const bf16_t* __restrict__ A, const bf16_t* __restrict__ Bt,
    const float* __restrict__ bias, float* __restrict__ C,
    int M, int N, int K)
{
  constexpr int LDA = 40;   // padded LDS stride (elements, 80B -> 16B aligned)
  constexpr int LDB = 40;
  __shared__ bf16_t sA[128 * LDA];
  __shared__ bf16_t sB[64 * LDB];

  const int tid  = threadIdx.x;
  const int lane = tid & 31;
  const int wave = tid >> 5;
  const int m0   = blockIdx.x * 128;
  const int n0   = blockIdx.y * 64;

  v8f acc[4] = {};

  const int arow  = tid >> 1;          // 0..127
  const int ahalf = (tid & 1) * 16;    // 0 or 16 (elements)
  const int brow  = (tid & 127) >> 1;  // 0..63
  const int bhalf = (tid & 1) * 16;
  const bool doB  = tid < 128;

  const int r  = lane & 15;
  const int hi = lane >> 4;            // 0 or 1

  for (int kt = 0; kt < K; kt += 32) {
    // --- stage A tile [128 x 32] and B^T tile [64 x 32] into LDS ---
    const bf16_t* gA = A + (size_t)(m0 + arow) * K + kt + ahalf;
    const bf16_t* gB = Bt + (size_t)(n0 + brow) * K + kt + bhalf;
#ifdef HAS_ASYNC_LDS
    async_copy16(gA,     &sA[arow * LDA + ahalf]);
    async_copy16(gA + 8, &sA[arow * LDA + ahalf + 8]);
    if (doB) {
      async_copy16(gB,     &sB[brow * LDB + bhalf]);
      async_copy16(gB + 8, &sB[brow * LDB + bhalf + 8]);
    }
#else
    *(v8bf*)(&sA[arow * LDA + ahalf])     = *(const v8bf*)(gA);
    *(v8bf*)(&sA[arow * LDA + ahalf + 8]) = *(const v8bf*)(gA + 8);
    if (doB) {
      *(v8bf*)(&sB[brow * LDB + bhalf])     = *(const v8bf*)(gB);
      *(v8bf*)(&sB[brow * LDB + bhalf + 8]) = *(const v8bf*)(gB + 8);
    }
#endif
    if (kt + 32 < K)  // hint next A tile toward the caches
      __builtin_prefetch(A + (size_t)(m0 + arow) * K + (kt + 32) + ahalf, 0, 1);
#ifdef HAS_ASYNC_LDS
    __builtin_amdgcn_s_wait_asynccnt(0);
#endif
    __syncthreads();

    // --- A fragment (ISA 16-bit A 16x32 layout):
    // lanes 0-15: row M=lane, K = 0..7 then 16..23 ; lanes 16-31: K = 8..15 then 24..31
    union { v16bf v; v8bf h[2]; } ua;
    const bf16_t* ap = &sA[(wave * 16 + r) * LDA + hi * 8];
    ua.h[0] = *(const v8bf*)(ap);
    ua.h[1] = *(const v8bf*)(ap + 16);

    // --- 4 B fragments (32x16, same K order as A: lanes 0-15 K=0..7,16..23; lanes 16-31 K=8..15,24..31)
    #pragma unroll
    for (int j = 0; j < 4; ++j) {
      union { v16bf v; v8bf h[2]; } ub;
      const bf16_t* bp = &sB[(j * 16 + r) * LDB + hi * 8];
      ub.h[0] = *(const v8bf*)(bp);
      ub.h[1] = *(const v8bf*)(bp + 16);
      acc[j] = __builtin_amdgcn_wmma_f32_16x16x32_bf16(
          false, ua.v, false, ub.v, (short)0, acc[j], false, false);
    }
    __syncthreads();
  }

  // --- store C (ISA 32-bit C/D layout: vgpr rr -> M = rr + 8*hi, lane&15 -> N)
  #pragma unroll
  for (int j = 0; j < 4; ++j) {
    const int col = n0 + j * 16 + r;
    const float bv = bias ? bias[col] : 0.0f;
    #pragma unroll
    for (int rr = 0; rr < 8; ++rr) {
      const int row = m0 + wave * 16 + rr + hi * 8;
      C[(size_t)row * N + col] = acc[j][rr] + bv;
    }
  }
}

// ---------------------------------------------------------------------------
// Packing / setup kernels
// ---------------------------------------------------------------------------
__global__ void pack_mem_kernel(const float* __restrict__ attn_mem,
                                const float* __restrict__ stop,
                                const int* __restrict__ msz,
                                bf16_t* __restrict__ mem)
{
  size_t idx = (size_t)blockIdx.x * blockDim.x + threadIdx.x;  // B*257*512
  int d  = idx % DD;
  int n  = (idx / DD) % NP1;
  int b  = idx / ((size_t)DD * NP1);
  float v = (n < 256) ? attn_mem[((size_t)b * 256 + n) * DD + d] : 0.0f;
  if (n == msz[b]) v += stop[d];
  mem[idx] = (bf16_t)v;
}

// W: [K,N] row-major -> Wt: [N,K] row-major (bf16)
__global__ void pack_wt_kernel(const float* __restrict__ W, bf16_t* __restrict__ Wt,
                               int K, int N)
{
  size_t idx = (size_t)blockIdx.x * blockDim.x + threadIdx.x;  // N*K
  int k = idx % K;
  int n = idx / K;
  Wt[idx] = (bf16_t)W[(size_t)k * N + n];
}

__global__ void pack_cvt_kernel(const float* __restrict__ W, bf16_t* __restrict__ O)
{
  size_t idx = (size_t)blockIdx.x * blockDim.x + threadIdx.x;
  O[idx] = (bf16_t)W[idx];
}

__global__ void pack_x_kernel(const float* __restrict__ lstm_in,
                              const float* __restrict__ init_i,
                              bf16_t* __restrict__ x)
{
  size_t idx = (size_t)blockIdx.x * blockDim.x + threadIdx.x;  // 1152*512
  int d  = idx & 511;
  int bq = idx >> 9;
  int t  = bq % T9;
  int b  = bq / T9;
  float v = (t == 0) ? init_i[d] : lstm_in[((size_t)b * 8 + (t - 1)) * DD + d];
  x[idx] = (bf16_t)v;
}

__global__ void init_state_kernel(const float* __restrict__ init_h,
                                  const float* __restrict__ init_c,
                                  bf16_t* __restrict__ hbf, float* __restrict__ c)
{
  int idx = blockIdx.x * blockDim.x + threadIdx.x;  // 128*512
  int hh = idx & 511;
  hbf[idx] = (bf16_t)init_h[hh];
  c[idx] = init_c[hh];
}

__global__ void bias_sum_kernel(const float* __restrict__ bih,
                                const float* __restrict__ bhh,
                                float* __restrict__ bias)
{
  int i = blockIdx.x * blockDim.x + threadIdx.x;  // 2048
  bias[i] = bih[i] + bhh[i];
}

// ---------------------------------------------------------------------------
// LSTM cell (torch gate order i,f,g,o)
// ---------------------------------------------------------------------------
__device__ __forceinline__ float sigf(float x) { return 1.0f / (1.0f + __expf(-x)); }

__global__ void lstm_cell_kernel(const float* __restrict__ xg,  // [1152,2048]
                                 const float* __restrict__ hg,  // [128,2048]
                                 float* __restrict__ c,         // [128,512]
                                 bf16_t* __restrict__ hbf,      // [128,512]
                                 bf16_t* __restrict__ qbf,      // [1152,512]
                                 int t)
{
  int idx = blockIdx.x * blockDim.x + threadIdx.x;  // b*512+h
  int b  = idx >> 9;
  int hh = idx & 511;
  const float* xr = xg + ((size_t)(b * T9 + t)) * G4H;
  const float* hr = hg + (size_t)b * G4H;
  float gi = xr[hh]        + hr[hh];
  float gf = xr[hh + 512]  + hr[hh + 512];
  float gg = xr[hh + 1024] + hr[hh + 1024];
  float go = xr[hh + 1536] + hr[hh + 1536];
  float cn = sigf(gf) * c[idx] + sigf(gi) * tanhf(gg);
  float hn = sigf(go) * tanhf(cn);
  c[idx] = cn;
  hbf[idx] = (bf16_t)hn;
  qbf[((size_t)(b * T9 + t)) * HH + hh] = (bf16_t)hn;
}

// ---------------------------------------------------------------------------
// score[bq,n] = sum_h v[h]*tanh(feat[b,n,h] + q[bq,h]); masked -> -1e18
// One block per (b,q); one wave per n (lanes stride h, shfl_xor reduce).
// ---------------------------------------------------------------------------
__global__ __launch_bounds__(256) void score_kernel(
    const float* __restrict__ feat,   // [B,257,512]
    const float* __restrict__ q,      // [1152,512]
    const float* __restrict__ vvec,   // [512]
    const int* __restrict__ msz,      // nullptr => no mask
    float* __restrict__ out)          // [1152,257]
{
  __shared__ float sq[HH];
  __shared__ float sv[HH];
  const int bq  = blockIdx.x;
  const int b   = bq / T9;
  const int tid = threadIdx.x;
  for (int i = tid; i < HH; i += 256) { sq[i] = q[(size_t)bq * HH + i]; sv[i] = vvec[i]; }
  __syncthreads();
  const int lane = tid & 31, wave = tid >> 5;
  const int limit = msz ? (msz[b] + 1) : NP1;
  const float* fb = feat + (size_t)b * NP1 * HH;
  for (int n = wave; n < NP1; n += 8) {
    float p;
    if (n < limit) {
      p = 0.0f;
      const float* fr = fb + (size_t)n * HH;
      for (int h = lane; h < HH; h += 32)
        p += sv[h] * tanhf(fr[h] + sq[h]);
      for (int off = 16; off; off >>= 1) p += __shfl_xor(p, off, 32);
    } else {
      p = -1e18f;
    }
    if (lane == 0) out[(size_t)bq * NP1 + n] = p;
  }
}

// ---------------------------------------------------------------------------
// softmax over n (257) then context: qout[bq,h] = sum_n p[n]*feat[b,n,h] (bf16)
// ---------------------------------------------------------------------------
__global__ __launch_bounds__(256) void softmax_ctx_kernel(
    const float* __restrict__ score,  // [1152,257]
    const float* __restrict__ feat,   // [B,257,512]
    bf16_t* __restrict__ qout)        // [1152,512]
{
  __shared__ float sp[NP1];
  __shared__ float red[256];
  const int bq = blockIdx.x, b = bq / T9, tid = threadIdx.x;

  float lmax = -1e30f;
  for (int n = tid; n < NP1; n += 256) {
    float s = score[(size_t)bq * NP1 + n];
    sp[n] = s;
    lmax = fmaxf(lmax, s);
  }
  red[tid] = lmax; __syncthreads();
  for (int s = 128; s; s >>= 1) { if (tid < s) red[tid] = fmaxf(red[tid], red[tid + s]); __syncthreads(); }
  const float mx = red[0]; __syncthreads();

  float lsum = 0.0f;
  for (int n = tid; n < NP1; n += 256) { float e = __expf(sp[n] - mx); sp[n] = e; lsum += e; }
  red[tid] = lsum; __syncthreads();
  for (int s = 128; s; s >>= 1) { if (tid < s) red[tid] += red[tid + s]; __syncthreads(); }
  const float inv = 1.0f / red[0]; __syncthreads();

  for (int n = tid; n < NP1; n += 256) sp[n] *= inv;
  __syncthreads();

  const float* fb = feat + (size_t)b * NP1 * HH;
  for (int h = tid; h < HH; h += 256) {
    float acc = 0.0f;
    for (int n = 0; n < NP1; ++n) acc += sp[n] * fb[(size_t)n * HH + h];
    qout[(size_t)bq * HH + h] = (bf16_t)acc;
  }
}

// ---------------------------------------------------------------------------
extern "C" void kernel_launch(void* const* d_in, const int* in_sizes, int n_in,
                              void* d_out, int out_size, void* d_ws, size_t ws_size,
                              hipStream_t stream) {
  const float* attn_mem  = (const float*)d_in[0];
  const float* lstm_in   = (const float*)d_in[1];
  const int*   mem_sizes = (const int*)d_in[2];
  const float* init_h    = (const float*)d_in[3];
  const float* init_c    = (const float*)d_in[4];
  const float* init_i    = (const float*)d_in[5];
  const float* stop      = (const float*)d_in[6];
  const float* w_ih      = (const float*)d_in[7];
  const float* w_hh      = (const float*)d_in[8];
  const float* b_ih      = (const float*)d_in[9];
  const float* b_hh      = (const float*)d_in[10];
  const float* attn_wm   = (const float*)d_in[11];
  const float* attn_wq   = (const float*)d_in[12];
  const float* attn_v    = (const float*)d_in[13];
  const float* hop_wm    = (const float*)d_in[14];
  const float* hop_wq    = (const float*)d_in[15];
  const float* hop_v     = (const float*)d_in[16];
  float* out = (float*)d_out;

  char* ws = (char*)d_ws;
  auto alloc = [&](size_t bytes) -> char* {
    char* p = ws;
    ws += (bytes + 255) & ~(size_t)255;
    return p;
  };

  const size_t Mfeat = (size_t)BB * NP1;          // 32896
  const size_t Mq    = (size_t)BB * T9;           // 1152

  bf16_t* mem_bf    = (bf16_t*)alloc(Mfeat * DD * 2);
  float*  attn_feat = (float*) alloc(Mfeat * HH * 4);
  float*  hop_feat  = (float*) alloc(Mfeat * HH * 4);
  bf16_t* attn_wm_t = (bf16_t*)alloc((size_t)DD * HH * 2);
  bf16_t* hop_wm_t  = (bf16_t*)alloc((size_t)DD * HH * 2);
  bf16_t* attn_wq_t = (bf16_t*)alloc((size_t)HH * HH * 2);
  bf16_t* hop_wq_t  = (bf16_t*)alloc((size_t)HH * HH * 2);
  bf16_t* w_ih_bf   = (bf16_t*)alloc((size_t)G4H * DD * 2);
  bf16_t* w_hh_bf   = (bf16_t*)alloc((size_t)G4H * HH * 2);
  float*  bias_sum  = (float*) alloc((size_t)G4H * 4);
  bf16_t* x_bf      = (bf16_t*)alloc(Mq * DD * 2);
  float*  xg        = (float*) alloc(Mq * G4H * 4);
  bf16_t* h_bf      = (bf16_t*)alloc((size_t)BB * HH * 2);
  float*  c_st      = (float*) alloc((size_t)BB * HH * 4);
  float*  hg        = (float*) alloc((size_t)BB * G4H * 4);
  bf16_t* q_bf      = (bf16_t*)alloc(Mq * HH * 2);
  float*  q2        = (float*) alloc(Mq * HH * 4);
  float*  score     = (float*) alloc(Mq * NP1 * 4);
  bf16_t* q2_bf     = (bf16_t*)alloc(Mq * HH * 2);
  float*  q3        = (float*) alloc(Mq * HH * 4);

  // ---- packing ----
  pack_mem_kernel<<<(Mfeat * DD) / 256, 256, 0, stream>>>(attn_mem, stop, mem_sizes, mem_bf);
  pack_wt_kernel<<<((size_t)DD * HH) / 256, 256, 0, stream>>>(attn_wm, attn_wm_t, DD, HH);
  pack_wt_kernel<<<((size_t)DD * HH) / 256, 256, 0, stream>>>(hop_wm, hop_wm_t, DD, HH);
  pack_wt_kernel<<<((size_t)HH * HH) / 256, 256, 0, stream>>>(attn_wq, attn_wq_t, HH, HH);
  pack_wt_kernel<<<((size_t)HH * HH) / 256, 256, 0, stream>>>(hop_wq, hop_wq_t, HH, HH);
  pack_cvt_kernel<<<((size_t)G4H * DD) / 256, 256, 0, stream>>>(w_ih, w_ih_bf);
  pack_cvt_kernel<<<((size_t)G4H * HH) / 256, 256, 0, stream>>>(w_hh, w_hh_bf);
  bias_sum_kernel<<<G4H / 256, 256, 0, stream>>>(b_ih, b_hh, bias_sum);
  pack_x_kernel<<<(Mq * DD) / 256, 256, 0, stream>>>(lstm_in, init_i, x_bf);
  init_state_kernel<<<((size_t)BB * HH) / 256, 256, 0, stream>>>(init_h, init_c, h_bf, c_st);

  // ---- feature GEMMs: [32896,512] x [512,512] ----
  {
    dim3 g(Mfeat / 128, HH / 64);
    gemm_bf16_kernel<<<g, 256, 0, stream>>>(mem_bf, attn_wm_t, nullptr, attn_feat,
                                            (int)Mfeat, HH, DD);
    gemm_bf16_kernel<<<g, 256, 0, stream>>>(mem_bf, hop_wm_t, nullptr, hop_feat,
                                            (int)Mfeat, HH, DD);
  }

  // ---- xg = x @ w_ih^T + (b_ih + b_hh): [1152,512] x [512,2048] ----
  {
    dim3 g(Mq / 128, G4H / 64);
    gemm_bf16_kernel<<<g, 256, 0, stream>>>(x_bf, w_ih_bf, bias_sum, xg,
                                            (int)Mq, G4H, DD);
  }

  // ---- LSTM: 9 sequential steps ----
  for (int t = 0; t < T9; ++t) {
    dim3 g(BB / 128, G4H / 64);
    gemm_bf16_kernel<<<g, 256, 0, stream>>>(h_bf, w_hh_bf, nullptr, hg,
                                            BB, G4H, HH);
    lstm_cell_kernel<<<((size_t)BB * HH) / 256, 256, 0, stream>>>(xg, hg, c_st, h_bf, q_bf, t);
  }

  // ---- hop attention (NHOP = 1) ----
  {
    dim3 g(Mq / 128, HH / 64);
    gemm_bf16_kernel<<<g, 256, 0, stream>>>(q_bf, hop_wq_t, nullptr, q2,
                                            (int)Mq, HH, HH);
  }
  score_kernel<<<(int)Mq, 256, 0, stream>>>(hop_feat, q2, hop_v, mem_sizes, score);
  softmax_ctx_kernel<<<(int)Mq, 256, 0, stream>>>(score, hop_feat, q2_bf);

  // ---- final pointer scores ----
  {
    dim3 g(Mq / 128, HH / 64);
    gemm_bf16_kernel<<<g, 256, 0, stream>>>(q2_bf, attn_wq_t, nullptr, q3,
                                            (int)Mq, HH, HH);
  }
  score_kernel<<<(int)Mq, 256, 0, stream>>>(attn_feat, q3, attn_v, nullptr, out);
}
